// SNN_36326833389728
// MI455X (gfx1250) — matrix operand order repaired and checked by reading, not compiled, not run on hardware
//
#include <hip/hip_runtime.h>

// ---------------- types for WMMA operands (wave32, gfx1250) ----------------
typedef _Float16 h16 __attribute__((ext_vector_type(16)));
typedef _Float16 h8v __attribute__((ext_vector_type(8)));
typedef float    f8v __attribute__((ext_vector_type(8)));

namespace {
constexpr int kT   = 100;
constexpr int kB   = 256;
constexpr int kIN  = 700, kINP = 704;   // pad K to multiple of 32
constexpr int kH   = 1024;
constexpr int kOUT = 20,  kOUTP = 32;   // pad N to multiple of 32
constexpr float kDT = 0.01f, kMinTau = 0.1f, kVth = 1.0f;

// ---- workspace layout (bytes); every block is a multiple of 256 ----
constexpr size_t OFF_SF = 0;
constexpr size_t SZ_SF  = (size_t)kT * kB * kINP * 2;      // s cast to f16, padded
constexpr size_t OFF_W1 = OFF_SF + SZ_SF;
constexpr size_t SZ_W1  = (size_t)kH * kINP * 2;
constexpr size_t OFF_W2 = OFF_W1 + SZ_W1;
constexpr size_t SZ_W2  = (size_t)kH * kH * 2;
constexpr size_t OFF_W3 = OFF_W2 + SZ_W2;
constexpr size_t SZ_W3  = (size_t)kOUTP * kH * 2;
constexpr size_t OFF_V1 = OFF_W3 + SZ_W3;
constexpr size_t SZ_V1  = (size_t)kB * kH * 4;
constexpr size_t OFF_V2 = OFF_V1 + SZ_V1;
constexpr size_t SZ_V2  = (size_t)kB * kH * 4;
constexpr size_t OFF_V3 = OFF_V2 + SZ_V2;
constexpr size_t SZ_V3  = (size_t)kB * kOUTP * 4;
constexpr size_t OFF_S1 = OFF_V3 + SZ_V3;
constexpr size_t SZ_S1  = (size_t)kB * kH * 2;
constexpr size_t OFF_S2 = OFF_S1 + SZ_S1;
} // namespace

// ---------------- fragment loaders per CDNA5 ISA 7.12.2 layouts ----------------
// A (16-bit, 16x32 MxK): lane m (0-15) holds K = k0+8*half..+7 in VGPR0-3 and
// K = k0+16+8*half..+7 in VGPR4-7 (half = lane>>4).
__device__ __forceinline__ h16 frag_a(const _Float16* __restrict__ base, int stride,
                                      int row, int k0, int half_) {
    const _Float16* p = base + (long)row * stride + k0 + 8 * half_;
    h8v lo = *(const h8v*)(p);
    h8v hi = *(const h8v*)(p + 16);
    h16 r;
#pragma unroll
    for (int i = 0; i < 8; ++i) { r[i] = lo[i]; r[i + 8] = hi[i]; }
    return r;
}

// B (16-bit, 32x16 KxN): lane n (0-15) holds K = k0+8*half..+7 in VGPR0-3 and
// K = k0+16+8*half..+7 in VGPR4-7 (half = lane>>4). B column n == row n of W (since cur = s@W^T).
__device__ __forceinline__ h16 frag_b(const _Float16* __restrict__ base, int stride,
                                      int col, int k0, int half_) {
    const _Float16* p = base + (long)col * stride + k0 + 8 * half_;
    h8v lo = *(const h8v*)(p);
    h8v hi = *(const h8v*)(p + 16);
    h16 r;
#pragma unroll
    for (int i = 0; i < 8; ++i) { r[i] = lo[i]; r[i + 8] = hi[i]; }
    return r;
}

// ---------------- fused GEMM (WMMA f16->f32) + LIF epilogue ----------------
// Grid: (N/128, 256/64), 256 threads (8 waves). Wave computes a 32x32 tile.
// cur = A[256,Kp] @ W[N,Kp]^T + bias; LIF: tau = min_tau + sigmoid(wtau),
// v_new = v + (DT/tau)*(cur - v); s = heaviside(v_new - Vth); v <- (1-s)*v_new.
__global__ __launch_bounds__(256, 1)
void snn_gemm_lif(const _Float16* __restrict__ A,   // [256, Kp]
                  const _Float16* __restrict__ W,   // [N, Kp] (padded)
                  const float* __restrict__ bias,   // [realN]
                  const float* __restrict__ wtau,   // [realN]
                  float* __restrict__ v,            // [256, N] state, updated in place
                  _Float16* __restrict__ spk,       // [256, N] f16 spikes (next A) or null
                  float* __restrict__ out_s,        // [256, realN] or null (layer 3)
                  float* __restrict__ out_v,        // [256, realN] or null
                  int N, int realN, int Kp) {
    const int lane  = threadIdx.x & 31;
    const int wid   = threadIdx.x >> 5;
    const int half_ = lane >> 4;
    const int idx   = lane & 15;
    const int wm    = wid & 1;
    const int wn    = wid >> 1;
    const int m0    = blockIdx.y * 64 + wm * 32;
    const int n0    = blockIdx.x * 128 + wn * 32;
    if (n0 >= N) return;  // wave-uniform: EXEC stays all-ones inside the K loop

    f8v acc[2][2];
#pragma unroll
    for (int mi = 0; mi < 2; ++mi)
#pragma unroll
        for (int ni = 0; ni < 2; ++ni)
#pragma unroll
            for (int e = 0; e < 8; ++e) acc[mi][ni][e] = 0.0f;

    for (int k0 = 0; k0 < Kp; k0 += 32) {
        if (k0 + 32 < Kp) {  // next K-slab into cache: emits global_prefetch_b8
            __builtin_prefetch(A + (long)(m0 + idx) * Kp + k0 + 32, 0, 3);
            __builtin_prefetch(W + (long)(n0 + idx) * Kp + k0 + 32, 0, 3);
        }
        h16 a0 = frag_a(A, Kp, m0 + idx,      k0, half_);
        h16 a1 = frag_a(A, Kp, m0 + 16 + idx, k0, half_);
        h16 b0 = frag_b(W, Kp, n0 + idx,      k0, half_);
        h16 b1 = frag_b(W, Kp, n0 + 16 + idx, k0, half_);
        acc[0][0] = __builtin_amdgcn_wmma_f32_16x16x32_f16(false, a0, false, b0,
                                                           (short)0, acc[0][0], false, false);
        acc[0][1] = __builtin_amdgcn_wmma_f32_16x16x32_f16(false, a0, false, b1,
                                                           (short)0, acc[0][1], false, false);
        acc[1][0] = __builtin_amdgcn_wmma_f32_16x16x32_f16(false, a1, false, b0,
                                                           (short)0, acc[1][0], false, false);
        acc[1][1] = __builtin_amdgcn_wmma_f32_16x16x32_f16(false, a1, false, b1,
                                                           (short)0, acc[1][1], false, false);
    }

    // D layout: lanes 0-15: VGPR r -> (M = r, N = lane); lanes 16-31: (M = 8+r, N = lane-16)
#pragma unroll
    for (int ni = 0; ni < 2; ++ni) {
        const int n = n0 + ni * 16 + idx;
        if (n >= realN) continue;  // skip N-padding columns (layer 3)
        const float bn  = bias[n];
        const float sg  = 1.0f / (1.0f + __expf(-wtau[n]));
        const float alp = kDT / (kMinTau + sg);
#pragma unroll
        for (int mi = 0; mi < 2; ++mi) {
#pragma unroll
            for (int r = 0; r < 8; ++r) {
                const int row = m0 + mi * 16 + 8 * half_ + r;
                const long vi = (long)row * N + n;
                const float cur  = acc[mi][ni][r] + bn;
                const float vold = v[vi];
                const float vnew = vold + alp * (cur - vold);   // VREST = 0
                const float s    = vnew > kVth ? 1.0f : 0.0f;
                v[vi] = (s > 0.0f) ? 0.0f : vnew;               // reset-to-zero
                if (spk)   spk[vi] = (_Float16)s;
                if (out_s) {
                    out_s[(long)row * realN + n] = s;
                    out_v[(long)row * realN + n] = vnew;        // pre-reset membrane
                }
            }
        }
    }
}

// ---------------- prep kernels ----------------
__global__ void k_cvt_pad(const float* __restrict__ src, _Float16* __restrict__ dst,
                          int rows_real, int cols_real, int rows_pad, int cols_pad) {
    long i = (long)blockIdx.x * blockDim.x + threadIdx.x;
    long n = (long)rows_pad * cols_pad;
    if (i >= n) return;
    int r = (int)(i / cols_pad);
    int c = (int)(i % cols_pad);
    float x = (r < rows_real && c < cols_real) ? src[(long)r * cols_real + c] : 0.0f;
    dst[i] = (_Float16)x;
}

__global__ void k_zero_f32(float* __restrict__ p, long n) {
    long i = (long)blockIdx.x * blockDim.x + threadIdx.x;
    if (i < n) p[i] = 0.0f;
}

// ---------------- host side ----------------
extern "C" void kernel_launch(void* const* d_in, const int* in_sizes, int n_in,
                              void* d_out, int out_size, void* d_ws, size_t ws_size,
                              hipStream_t stream) {
    const float* s   = (const float*)d_in[0];
    const float* W1  = (const float*)d_in[1];
    const float* b1  = (const float*)d_in[2];
    const float* wt1 = (const float*)d_in[3];
    const float* W2  = (const float*)d_in[4];
    const float* b2  = (const float*)d_in[5];
    const float* wt2 = (const float*)d_in[6];
    const float* W3  = (const float*)d_in[7];
    const float* b3  = (const float*)d_in[8];
    const float* wt3 = (const float*)d_in[9];

    float* outS = (float*)d_out;                             // [T,B,20] spikes
    float* outV = (float*)d_out + (long)kT * kB * kOUT;      // [T,B,20] pre-reset v

    char* ws = (char*)d_ws;
    _Float16* SF  = (_Float16*)(ws + OFF_SF);
    _Float16* W1F = (_Float16*)(ws + OFF_W1);
    _Float16* W2F = (_Float16*)(ws + OFF_W2);
    _Float16* W3F = (_Float16*)(ws + OFF_W3);
    float*    V1  = (float*)(ws + OFF_V1);
    float*    V3  = (float*)(ws + OFF_V3);
    float*    V2  = (float*)(ws + OFF_V2);
    _Float16* S1  = (_Float16*)(ws + OFF_S1);
    _Float16* S2  = (_Float16*)(ws + OFF_S2);

    auto blocks = [](long n) { return (unsigned)((n + 255) / 256); };

    // cast + pad inputs/weights to f16 WMMA-friendly layouts
    k_cvt_pad<<<blocks((long)kT * kB * kINP), 256, 0, stream>>>(s, SF, kT * kB, kIN, kT * kB, kINP);
    k_cvt_pad<<<blocks((long)kH * kINP), 256, 0, stream>>>(W1, W1F, kH, kIN, kH, kINP);
    k_cvt_pad<<<blocks((long)kH * kH), 256, 0, stream>>>(W2, W2F, kH, kH, kH, kH);
    k_cvt_pad<<<blocks((long)kOUTP * kH), 256, 0, stream>>>(W3, W3F, kOUT, kH, kOUTP, kH);
    // zero membrane state (V1,V2,V3 are contiguous in ws)
    const long vcount = (long)(SZ_V1 + SZ_V2 + SZ_V3) / 4;
    k_zero_f32<<<blocks(vcount), 256, 0, stream>>>(V1, vcount);

    const dim3 gridH(kH / 128, kB / 64);   // (8,4) for N=1024
    const dim3 gridO(1, kB / 64);          // (1,4) for N=32 (layer 3)

    for (int t = 0; t < kT; ++t) {
        // layer 1: s_t[256,704] @ W1^T -> LIF -> S1 spikes (f16)
        snn_gemm_lif<<<gridH, 256, 0, stream>>>(SF + (long)t * kB * kINP, W1F, b1, wt1,
                                                V1, S1, nullptr, nullptr, kH, kH, kINP);
        // layer 2: S1 @ W2^T -> LIF -> S2 spikes (f16)
        snn_gemm_lif<<<gridH, 256, 0, stream>>>(S1, W2F, b2, wt2,
                                                V2, S2, nullptr, nullptr, kH, kH, kH);
        // layer 3: S2 @ W3^T -> LIF -> write spikes + pre-reset membrane to d_out
        snn_gemm_lif<<<gridO, 256, 0, stream>>>(S2, W3F, b3, wt3,
                                                V3, nullptr,
                                                outS + (long)t * kB * kOUT,
                                                outV + (long)t * kB * kOUT,
                                                kOUTP, kOUT, kH);
    }
    (void)in_sizes; (void)n_in; (void)out_size; (void)ws_size;
}
